// AttentionTimeSeriesTransformer_67499706024714
// MI455X (gfx1250) — matrix operand order repaired and checked by reading, not compiled, not run on hardware
//
#include <hip/hip_runtime.h>

// ---------------------------------------------------------------------------
// Transformer encoder forward for MI455X (gfx1250), wave32 + WMMA f16->f32.
// Compute-bound workload (~714 GFLOP vs ~50us of HBM time at 23.3 TB/s), so
// everything runs through v_wmma_f32_16x16x32_f16. GEMM uses 128x128 block
// tiles / 32x64 wave tiles (16 WMMAs per barrier pair) for WMMA-issue density.
// Flash-attention (online softmax) avoids the 256 MiB score tensor. Weights are
// converted to f16 once per launch; activations convert while staging to LDS.
// ---------------------------------------------------------------------------

#define DMODEL 512
#define NHEAD  8
#define NLAYER 6
#define FFDIM  2048
#define SEQ    512
#define BATCH  32
#define INDIM  64
#define HDIM   64
#define MROWS  (BATCH * SEQ)   // 16384

typedef __attribute__((ext_vector_type(16))) _Float16 v16h;
typedef __attribute__((ext_vector_type(8)))  float    v8f;

union F16x16 {
  v16h  v;
  uint4 u[2];
  _Float16 h[16];
};

static __device__ __forceinline__ v8f wmma_f16(const F16x16 &a, const F16x16 &b, v8f c) {
  // D = A(16x32 f16) * B(32x16 f16) + C(16x16 f32)
  return __builtin_amdgcn_wmma_f32_16x16x32_f16(
      /*neg_a=*/false, a.v, /*neg_b=*/false, b.v,
      /*c_mod=*/(short)0, c, /*reuse_a=*/false, /*reuse_b=*/false);
}

// ------------------------------ weight convert ------------------------------
__global__ __launch_bounds__(256) void f32_to_f16_kernel(const float* __restrict__ in,
                                                         _Float16* __restrict__ out, int n) {
  int i = blockIdx.x * 256 + threadIdx.x;
  if (i < n) out[i] = (_Float16)in[i];
}

// ------------------------------ generic GEMM --------------------------------
// C[M x N] = A[M x K](f32, lda) @ W[N x K](f16)^T + bias[N]
// flags: 0 = plain, 1 = ReLU, 2 = input-proj (scale by sqrt(D) + pos-encoding)
// Grid: (N/128, ceil(M/128)), block 256 (8 waves).
// Wave tile 32x64 (2x4 WMMA accumulators), BK=64 (two K=32 sub-steps).
__global__ __launch_bounds__(256) void gemm_kernel(const float* __restrict__ A, int lda,
                                                   const _Float16* __restrict__ W,
                                                   const float* __restrict__ bias,
                                                   float* __restrict__ C, int ldc,
                                                   int M, int K, int flags) {
  __shared__ __align__(16) _Float16 lds_a[128 * 64];  // 16 KB
  __shared__ __align__(16) _Float16 lds_w[128 * 64];  // 16 KB

  const int bm = blockIdx.y * 128;
  const int bn = blockIdx.x * 128;
  const int wave = threadIdx.x >> 5;
  const int lane = threadIdx.x & 31;
  const int lr = lane & 15;       // column / row within 16
  const int hi = lane >> 4;       // lane-group (0 or 1)
  const int wm = wave & 3;        // 0..3 -> 32-row strip
  const int wn = wave >> 2;       // 0..1 -> 64-col strip

  v8f zero = {};
  v8f acc[2][4];
  #pragma unroll
  for (int mi = 0; mi < 2; ++mi)
    #pragma unroll
    for (int ni = 0; ni < 4; ++ni) acc[mi][ni] = zero;

  for (int k0 = 0; k0 < K; k0 += 64) {
    // ---- stage A tile (128x64 f32 -> f16): 32 elems per thread ----
    {
      int r  = threadIdx.x >> 1;          // 0..127
      int c0 = (threadIdx.x & 1) << 5;    // 0 or 32
      int gm = bm + r; if (gm >= M) gm = M - 1;
      const float* ap = A + (size_t)gm * lda + k0 + c0;
      _Float16 tmp[32] __attribute__((aligned(16)));
      #pragma unroll
      for (int j = 0; j < 32; j += 4) {
        float4 f = *reinterpret_cast<const float4*>(ap + j);
        tmp[j + 0] = (_Float16)f.x; tmp[j + 1] = (_Float16)f.y;
        tmp[j + 2] = (_Float16)f.z; tmp[j + 3] = (_Float16)f.w;
      }
      #pragma unroll
      for (int j = 0; j < 32; j += 8)
        *reinterpret_cast<uint4*>(&lds_a[r * 64 + c0 + j]) =
            *reinterpret_cast<const uint4*>(&tmp[j]);
    }
    // ---- stage W tile (128x64 f16): 32 halves per thread ----
    {
      int r  = threadIdx.x >> 1;          // 0..127
      int c0 = (threadIdx.x & 1) << 5;    // 0 or 32
      int gn = bn + r;
      const uint4* wp = reinterpret_cast<const uint4*>(W + (size_t)gn * K + k0 + c0);
      uint4* dp = reinterpret_cast<uint4*>(&lds_w[r * 64 + c0]);
      #pragma unroll
      for (int j = 0; j < 4; ++j) dp[j] = wp[j];
    }
    __syncthreads();

    // ---- 2 K=32 sub-steps, 8 WMMAs each (ISA 7.12.2 fragment layouts) ----
    #pragma unroll
    for (int ks = 0; ks < 64; ks += 32) {
      F16x16 af[2];
      #pragma unroll
      for (int mi = 0; mi < 2; ++mi) {
        const _Float16* p = &lds_a[(wm * 32 + mi * 16 + lr) * 64 + ks + hi * 8];
        af[mi].u[0] = *reinterpret_cast<const uint4*>(p);
        af[mi].u[1] = *reinterpret_cast<const uint4*>(p + 16);
      }
      F16x16 bf[4];
      #pragma unroll
      for (int ni = 0; ni < 4; ++ni) {
        const _Float16* p = &lds_w[(wn * 64 + ni * 16 + lr) * 64 + ks + hi * 8];
        bf[ni].u[0] = *reinterpret_cast<const uint4*>(p);
        bf[ni].u[1] = *reinterpret_cast<const uint4*>(p + 16);
      }
      #pragma unroll
      for (int mi = 0; mi < 2; ++mi)
        #pragma unroll
        for (int ni = 0; ni < 4; ++ni)
          acc[mi][ni] = wmma_f16(af[mi], bf[ni], acc[mi][ni]);
    }
    __syncthreads();
  }

  // ---- epilogue ----
  #pragma unroll
  for (int mi = 0; mi < 2; ++mi) {
    #pragma unroll
    for (int ni = 0; ni < 4; ++ni) {
      #pragma unroll
      for (int v = 0; v < 8; ++v) {
        int gm = bm + wm * 32 + mi * 16 + v + 8 * hi;
        int gn = bn + wn * 64 + ni * 16 + lr;
        if (gm < M) {
          float val = acc[mi][ni][v] + bias[gn];
          if (flags == 1) val = fmaxf(val, 0.0f);
          if (flags == 2) {
            val *= 22.627416997969522f;                 // sqrt(512)
            int s  = gm & (SEQ - 1);
            int i2 = gn & ~1;
            float freq = __expf((float)i2 * (-9.210340371976184f / 512.0f));
            float ang  = (float)s * freq;
            val += (gn & 1) ? __cosf(ang) : __sinf(ang);
          }
          C[(size_t)gm * ldc + gn] = val;
        }
      }
    }
  }
}

// --------------------------- flash attention -------------------------------
// qkv: [B*S, 3*D] f32.  out: [B*S, D] f32.
// Grid (S/128, H, B), block 256 (8 waves); wave owns 16 query rows.
__global__ __launch_bounds__(256) void attn_kernel(const float* __restrict__ qkv,
                                                   float* __restrict__ out) {
  __shared__ __align__(16) _Float16 lds_k[32 * 64];    // K tile  (rows x HD)   4 KB
  __shared__ __align__(16) _Float16 lds_vt[64 * 32];   // V tile transposed     4 KB
  __shared__ __align__(16) _Float16 lds_p[8][16 * 32]; // per-wave P staging    8 KB

  const int b  = blockIdx.z;
  const int h  = blockIdx.y;
  const int q0 = blockIdx.x * 128;
  const int wave = threadIdx.x >> 5;
  const int lane = threadIdx.x & 31;
  const int lr = lane & 15;
  const int hi = lane >> 4;
  const int RS = 3 * DMODEL;   // qkv row stride
  const float scale = 0.125f;  // 1/sqrt(64)

  // Q a-fragments held in registers for the whole K loop (2 frags: K=64)
  F16x16 qa[2];
  {
    int qrow = b * SEQ + q0 + wave * 16 + lr;
    const float* qp = qkv + (size_t)qrow * RS + h * HDIM;
    #pragma unroll
    for (int st = 0; st < 2; ++st) {
      int kb = 32 * st + hi * 8;
      #pragma unroll
      for (int j = 0; j < 8; ++j) {
        qa[st].h[j]     = (_Float16)qp[kb + j];
        qa[st].h[8 + j] = (_Float16)qp[kb + 16 + j];
      }
    }
  }

  float m[8], lsum[8];
  v8f zero = {};
  v8f o[4];
  #pragma unroll
  for (int v = 0; v < 8; ++v) { m[v] = -3.0e38f; lsum[v] = 0.0f; }
  #pragma unroll
  for (int t = 0; t < 4; ++t) o[t] = zero;

  for (int kt = 0; kt < SEQ; kt += 32) {
    // cooperative staging of K (row-major f16) and V (transposed f16)
    {
      int r  = threadIdx.x >> 3;          // 0..31
      int c0 = (threadIdx.x & 7) << 3;    // 0..56
      const float* kp = qkv + (size_t)(b * SEQ + kt + r) * RS + DMODEL + h * HDIM + c0;
      _Float16 tmp[8] __attribute__((aligned(16)));
      #pragma unroll
      for (int j = 0; j < 8; ++j) tmp[j] = (_Float16)kp[j];
      *reinterpret_cast<uint4*>(&lds_k[r * 64 + c0]) = *reinterpret_cast<const uint4*>(&tmp[0]);
      const float* vp = qkv + (size_t)(b * SEQ + kt + r) * RS + 2 * DMODEL + h * HDIM + c0;
      #pragma unroll
      for (int j = 0; j < 8; ++j) lds_vt[(c0 + j) * 32 + r] = (_Float16)vp[j];
    }
    __syncthreads();

    // scores S = Q @ K^T for 32 key rows -> two 16x16 f32 tiles
    v8f s0 = {}, s1 = {};
    #pragma unroll
    for (int st = 0; st < 2; ++st) {
      F16x16 b0, b1;
      const _Float16* p0 = &lds_k[(0  + lr) * 64 + 32 * st + hi * 8];
      const _Float16* p1 = &lds_k[(16 + lr) * 64 + 32 * st + hi * 8];
      b0.u[0] = *reinterpret_cast<const uint4*>(p0);
      b0.u[1] = *reinterpret_cast<const uint4*>(p0 + 16);
      b1.u[0] = *reinterpret_cast<const uint4*>(p1);
      b1.u[1] = *reinterpret_cast<const uint4*>(p1 + 16);
      s0 = wmma_f16(qa[st], b0, s0);
      s1 = wmma_f16(qa[st], b1, s1);
    }

    // online softmax update (per output row = v + 8*hi; cols across 16 lanes)
    #pragma unroll
    for (int v = 0; v < 8; ++v) {
      float x0 = s0[v] * scale, x1 = s1[v] * scale;
      float t = fmaxf(x0, x1);
      t = fmaxf(t, __shfl_xor(t, 1, 32));
      t = fmaxf(t, __shfl_xor(t, 2, 32));
      t = fmaxf(t, __shfl_xor(t, 4, 32));
      t = fmaxf(t, __shfl_xor(t, 8, 32));
      float mn = fmaxf(m[v], t);
      float al = __expf(m[v] - mn);
      float p0 = __expf(x0 - mn);
      float p1 = __expf(x1 - mn);
      float rs = p0 + p1;
      rs += __shfl_xor(rs, 1, 32);
      rs += __shfl_xor(rs, 2, 32);
      rs += __shfl_xor(rs, 4, 32);
      rs += __shfl_xor(rs, 8, 32);
      m[v]    = mn;
      lsum[v] = lsum[v] * al + rs;
      #pragma unroll
      for (int t4 = 0; t4 < 4; ++t4) o[t4][v] = o[t4][v] * al;
      // C-layout -> LDS so it can be reloaded in A-layout
      lds_p[wave][(v + 8 * hi) * 32 + lr]      = (_Float16)p0;
      lds_p[wave][(v + 8 * hi) * 32 + 16 + lr] = (_Float16)p1;
    }
    asm volatile("s_wait_dscnt 0" ::: "memory");

    // O += P(16x32) @ V(32x64)
    F16x16 pa;
    const _Float16* pp = &lds_p[wave][lr * 32 + hi * 8];
    pa.u[0] = *reinterpret_cast<const uint4*>(pp);
    pa.u[1] = *reinterpret_cast<const uint4*>(pp + 16);
    #pragma unroll
    for (int t4 = 0; t4 < 4; ++t4) {
      F16x16 vb;
      const _Float16* vp2 = &lds_vt[(16 * t4 + lr) * 32 + hi * 8];
      vb.u[0] = *reinterpret_cast<const uint4*>(vp2);
      vb.u[1] = *reinterpret_cast<const uint4*>(vp2 + 16);
      o[t4] = wmma_f16(pa, vb, o[t4]);
    }
    __syncthreads();
  }

  // normalize and write
  #pragma unroll
  for (int t4 = 0; t4 < 4; ++t4) {
    #pragma unroll
    for (int v = 0; v < 8; ++v) {
      int row = q0 + wave * 16 + v + 8 * hi;
      int col = h * HDIM + 16 * t4 + lr;
      out[(size_t)(b * SEQ + row) * DMODEL + col] = o[t4][v] / lsum[v];
    }
  }
}

// --------------------------- residual + layernorm ---------------------------
// x = LN(x + branch) * g + b ; one wave per row of 512. Grid: MROWS/8, 256.
__global__ __launch_bounds__(256) void ln_kernel(float* __restrict__ x,
                                                 const float* __restrict__ branch,
                                                 const float* __restrict__ g,
                                                 const float* __restrict__ bta) {
  const int wave = threadIdx.x >> 5;
  const int lane = threadIdx.x & 31;
  const size_t row = (size_t)blockIdx.x * 8 + wave;
  float* xr = x + row * DMODEL;
  const float* rr = branch + row * DMODEL;
  float val[16];
  float s = 0.0f, ss = 0.0f;
  #pragma unroll
  for (int i = 0; i < 16; ++i) {
    int c = lane + 32 * i;
    float t = xr[c] + rr[c];
    val[i] = t; s += t; ss += t * t;
  }
  #pragma unroll
  for (int msk = 1; msk < 32; msk <<= 1) {
    s  += __shfl_xor(s,  msk, 32);
    ss += __shfl_xor(ss, msk, 32);
  }
  const float mean = s * (1.0f / 512.0f);
  const float var  = ss * (1.0f / 512.0f) - mean * mean;
  const float rstd = rsqrtf(var + 1e-5f);
  #pragma unroll
  for (int i = 0; i < 16; ++i) {
    int c = lane + 32 * i;
    xr[c] = (val[i] - mean) * rstd * g[c] + bta[c];
  }
}

// ------------------------------- head (N=1) --------------------------------
__global__ __launch_bounds__(256) void head2_kernel(const float* __restrict__ hbuf,
                                                    const float* __restrict__ w2,
                                                    const float* __restrict__ b2,
                                                    float* __restrict__ out) {
  __shared__ float red[8];
  const int b = blockIdx.x;
  float s = 0.0f;
  for (int j = threadIdx.x; j < FFDIM; j += 256) s += hbuf[b * FFDIM + j] * w2[j];
  #pragma unroll
  for (int msk = 1; msk < 32; msk <<= 1) s += __shfl_xor(s, msk, 32);
  if ((threadIdx.x & 31) == 0) red[threadIdx.x >> 5] = s;
  __syncthreads();
  if (threadIdx.x == 0) {
    float t = 0.0f;
    #pragma unroll
    for (int w = 0; w < 8; ++w) t += red[w];
    out[b] = t + b2[0];
  }
}

// ------------------------------- launcher ----------------------------------
extern "C" void kernel_launch(void* const* d_in, const int* in_sizes, int n_in,
                              void* d_out, int out_size, void* d_ws, size_t ws_size,
                              hipStream_t stream) {
  (void)in_sizes; (void)n_in; (void)out_size; (void)ws_size;
  const float* src       = (const float*)d_in[0];
  const float* inp_w     = (const float*)d_in[1];
  const float* inp_b     = (const float*)d_in[2];
  const float* in_proj_w = (const float*)d_in[3];
  const float* in_proj_b = (const float*)d_in[4];
  const float* out_w     = (const float*)d_in[5];
  const float* out_b     = (const float*)d_in[6];
  const float* ff_w1     = (const float*)d_in[7];
  const float* ff_b1     = (const float*)d_in[8];
  const float* ff_w2     = (const float*)d_in[9];
  const float* ff_b2     = (const float*)d_in[10];
  const float* ln1_g     = (const float*)d_in[11];
  const float* ln1_b     = (const float*)d_in[12];
  const float* ln2_g     = (const float*)d_in[13];
  const float* ln2_b     = (const float*)d_in[14];
  const float* op_w1     = (const float*)d_in[15];
  const float* op_b1     = (const float*)d_in[16];
  const float* op_w2     = (const float*)d_in[17];
  const float* op_b2     = (const float*)d_in[18];
  float* outp = (float*)d_out;

  // workspace layout (~262 MiB total)
  char* ws = (char*)d_ws;
  float* xf    = (float*)ws; ws += (size_t)MROWS * DMODEL * 4;   // activations
  float* buf1  = (float*)ws; ws += (size_t)MROWS * FFDIM  * 4;   // qkv / ff hidden
  float* bufA  = (float*)ws; ws += (size_t)MROWS * DMODEL * 4;   // attention out
  float* bufB  = (float*)ws; ws += (size_t)MROWS * DMODEL * 4;   // branch out
  float* headh = (float*)ws; ws += (size_t)BATCH * FFDIM  * 4;   // head hidden
  _Float16* w_inp = (_Float16*)ws;
  _Float16* w_qkv = w_inp + (size_t)DMODEL * INDIM;
  _Float16* w_out = w_qkv + (size_t)NLAYER * 3 * DMODEL * DMODEL;
  _Float16* w_ff1 = w_out + (size_t)NLAYER * DMODEL * DMODEL;
  _Float16* w_ff2 = w_ff1 + (size_t)NLAYER * FFDIM * DMODEL;
  _Float16* w_op1 = w_ff2 + (size_t)NLAYER * DMODEL * FFDIM;

  auto cvt = [&](const float* s, _Float16* d, size_t n) {
    f32_to_f16_kernel<<<dim3((unsigned)((n + 255) / 256)), dim3(256), 0, stream>>>(s, d, (int)n);
  };
  cvt(inp_w,     w_inp, (size_t)DMODEL * INDIM);
  cvt(in_proj_w, w_qkv, (size_t)NLAYER * 3 * DMODEL * DMODEL);
  cvt(out_w,     w_out, (size_t)NLAYER * DMODEL * DMODEL);
  cvt(ff_w1,     w_ff1, (size_t)NLAYER * FFDIM * DMODEL);
  cvt(ff_w2,     w_ff2, (size_t)NLAYER * DMODEL * FFDIM);
  cvt(op_w1,     w_op1, (size_t)FFDIM * DMODEL);

  const int M = MROWS;
  // input projection: x = (src @ inp_w^T + b) * sqrt(D) + pos_enc
  gemm_kernel<<<dim3(DMODEL / 128, M / 128), 256, 0, stream>>>(
      src, INDIM, w_inp, inp_b, xf, DMODEL, M, INDIM, 2);

  for (int i = 0; i < NLAYER; ++i) {
    // qkv projection
    gemm_kernel<<<dim3(3 * DMODEL / 128, M / 128), 256, 0, stream>>>(
        xf, DMODEL, w_qkv + (size_t)i * 3 * DMODEL * DMODEL, in_proj_b + i * 3 * DMODEL,
        buf1, 3 * DMODEL, M, DMODEL, 0);
    // flash attention
    attn_kernel<<<dim3(SEQ / 128, NHEAD, BATCH), 256, 0, stream>>>(buf1, bufA);
    // out projection
    gemm_kernel<<<dim3(DMODEL / 128, M / 128), 256, 0, stream>>>(
        bufA, DMODEL, w_out + (size_t)i * DMODEL * DMODEL, out_b + i * DMODEL,
        bufB, DMODEL, M, DMODEL, 0);
    // x = LN(x + attn)
    ln_kernel<<<dim3(M / 8), 256, 0, stream>>>(xf, bufB, ln1_g + i * DMODEL, ln1_b + i * DMODEL);
    // ff1 (ReLU)
    gemm_kernel<<<dim3(FFDIM / 128, M / 128), 256, 0, stream>>>(
        xf, DMODEL, w_ff1 + (size_t)i * FFDIM * DMODEL, ff_b1 + i * FFDIM,
        buf1, FFDIM, M, DMODEL, 1);
    // ff2
    gemm_kernel<<<dim3(DMODEL / 128, M / 128), 256, 0, stream>>>(
        buf1, FFDIM, w_ff2 + (size_t)i * DMODEL * FFDIM, ff_b2 + i * DMODEL,
        bufB, DMODEL, M, FFDIM, 0);
    // x = LN(x + ff)
    ln_kernel<<<dim3(M / 8), 256, 0, stream>>>(xf, bufB, ln2_g + i * DMODEL, ln2_b + i * DMODEL);
  }

  // head: last token rows (b, S-1) via stride trick; M=32 with in-kernel clamp
  gemm_kernel<<<dim3(FFDIM / 128, 1), 256, 0, stream>>>(
      xf + (size_t)(SEQ - 1) * DMODEL, SEQ * DMODEL, w_op1, op_b1,
      headh, FFDIM, BATCH, DMODEL, 1);
  head2_kernel<<<dim3(BATCH), 256, 0, stream>>>(headh, op_w2, op_b2, outp);
}
